// Predictor_27693949124933
// MI455X (gfx1250) — compile-verified, matrix operand order checked by reading, not run on hardware
//
#include <hip/hip_runtime.h>
#include <hip/hip_bf16.h>
#include <math.h>

// ---------------- problem constants (match reference) ----------------
#define NIMG        8
#define HH          100
#define WW          152
#define HW_         (HH * WW)            // 15200
#define TOK         256
#define PRE_TOPN    1000
#define POST_TOPN   100
#define NMS_TH      0.6f
#define PRE_TH      0.05f
#define IMG_H_      800.0f
#define IMG_W_      1216.0f
#define DWH_CLIP_   4.135166556742356f   // log(1000/16)
#define LOG2E_      1.4426950408889634f

typedef __attribute__((ext_vector_type(16))) _Float16 v16h;
typedef __attribute__((ext_vector_type(8)))  float    v8f;

// Fast sigmoid using raw HW transcendentals:
//   sigmoid(x) = rcp(1 + exp2(-x*log2(e)))
// 4 VALU ops (mul, v_exp_f32, add, v_rcp_f32) instead of the ~15-op
// IEEE-correct division sequence the compiler emits for "1.0f/x".
__device__ __forceinline__ float sigf(float x) {
    const float e = __builtin_amdgcn_exp2f(x * -LOG2E_);
    return __builtin_amdgcn_rcpf(1.0f + e);
}

__device__ __forceinline__ float frcp(float x) {
    return __builtin_amdgcn_rcpf(x);
}

// =====================================================================
// Kernel 1: scores[r] = mean_k sigmoid(logits[r,k]) fused with
//           masked[r] = (score > 0.05) ? score * sigmoid(ctr[r]) : 0
// One wave32 per 16-row tile; WMMA f32_16x16x32_f16 with all-ones B
// computes 16 row-sums per instruction (8 WMMAs cover T=256).
// A-matrix lane mapping (ISA 7.12.2, 16-bit A 16x32):
//   lane m    (m<16): row m, K in {0..7, 16..23}
//   lane m+16       : row m, K in {8..15, 24..31}
// Intra-lane K order is irrelevant since B == ones.
// =====================================================================
__global__ __launch_bounds__(256)
void scores_wmma_kernel(const float* __restrict__ logits,   // [N*HW, 256]
                        const float* __restrict__ ctr,      // [N*HW]
                        float* __restrict__ masked)         // [N*HW]
{
    const int wave = threadIdx.x >> 5;
    const int lane = threadIdx.x & 31;
    const int tile = blockIdx.x * 8 + wave;
    const int row0 = tile * 16;
    if (row0 >= NIMG * HW_) return;                 // uniform per wave

    const int m  = lane & 15;
    const int hi = lane >> 4;                       // K half selector
    const float* rp = logits + (size_t)(row0 + m) * TOK + hi * 8;

    v16h b;
#pragma unroll
    for (int i = 0; i < 16; ++i) b[i] = (_Float16)1.0f;

    v8f c = {};
#pragma unroll
    for (int kb = 0; kb < TOK; kb += 32) {
        float4 x0 = *(const float4*)(rp + kb + 0);
        float4 x1 = *(const float4*)(rp + kb + 4);
        float4 x2 = *(const float4*)(rp + kb + 16);
        float4 x3 = *(const float4*)(rp + kb + 20);
        v16h a;
        a[0]  = (_Float16)sigf(x0.x);  a[1]  = (_Float16)sigf(x0.y);
        a[2]  = (_Float16)sigf(x0.z);  a[3]  = (_Float16)sigf(x0.w);
        a[4]  = (_Float16)sigf(x1.x);  a[5]  = (_Float16)sigf(x1.y);
        a[6]  = (_Float16)sigf(x1.z);  a[7]  = (_Float16)sigf(x1.w);
        a[8]  = (_Float16)sigf(x2.x);  a[9]  = (_Float16)sigf(x2.y);
        a[10] = (_Float16)sigf(x2.z);  a[11] = (_Float16)sigf(x2.w);
        a[12] = (_Float16)sigf(x3.x);  a[13] = (_Float16)sigf(x3.y);
        a[14] = (_Float16)sigf(x3.z);  a[15] = (_Float16)sigf(x3.w);
        // D[m,n] = sum_k A[m,k]  (accumulated across kb chunks)
        c = __builtin_amdgcn_wmma_f32_16x16x32_f16(
                false, a, false, b, (short)0, c, false, false);
    }

    // D layout: lane l, VGPR g -> row (g + 8*(l>>4)); all N columns equal.
    // lane 0 emits rows 0..7, lane 16 emits rows 8..15.
    if (lane == 0 || lane == 16) {
        const int rbase = row0 + hi * 8;
#pragma unroll
        for (int g = 0; g < 8; ++g) {
            const int r = rbase + g;
            const float s = c[g] * (1.0f / (float)TOK);       // mean over tokens
            const float ct = sigf(ctr[r]);
            masked[r] = (s > PRE_TH) ? s * ct : 0.0f;
        }
    }
}

// =====================================================================
// Kernel 2: per-image top-1000 select via float-bit radix histogram.
// masked values are >= 0 so the raw IEEE bits are monotone; bin on the
// top 10 bits. One 1024-thread workgroup per image.
// =====================================================================
__global__ __launch_bounds__(1024)
void topk_select_kernel(const float* __restrict__ masked,   // [N*HW]
                        int* __restrict__ sel_idx)          // [N*PRE_TOPN]
{
    __shared__ unsigned hist[1024];
    __shared__ int s_cut, s_high, highCnt, edgeCnt;

    const int n   = blockIdx.x;
    const int tid = threadIdx.x;
    const float* mp = masked + (size_t)n * HW_;

    hist[tid] = 0u;
    if (tid < PRE_TOPN) sel_idx[n * PRE_TOPN + tid] = -1;
    __syncthreads();

    for (int hw = tid; hw < HW_; hw += 1024) {
        const float v = mp[hw];
        if (v > 0.0f)
            atomicAdd(&hist[__float_as_uint(v) >> 22], 1u);
    }
    __syncthreads();

    if (tid == 0) {
        int acc = 0, cut = -1;
        for (int bin = 1023; bin >= 0; --bin) {
            const int cb = (int)hist[bin];
            if (acc + cb >= PRE_TOPN) { cut = bin; break; }
            acc += cb;
        }
        s_cut = cut; s_high = acc; highCnt = 0; edgeCnt = 0;
    }
    __syncthreads();

    const int cut = s_cut, nhigh = s_high;
    for (int hw = tid; hw < HW_; hw += 1024) {
        const float v = mp[hw];
        if (v <= 0.0f) continue;
        const int bin = (int)(__float_as_uint(v) >> 22);
        if (bin > cut) {
            const int slot = atomicAdd(&highCnt, 1);        // < PRE_TOPN by scan
            sel_idx[n * PRE_TOPN + slot] = hw;
        } else if (bin == cut) {
            const int slot = nhigh + atomicAdd(&edgeCnt, 1);
            if (slot < PRE_TOPN) sel_idx[n * PRE_TOPN + slot] = hw;
        }
    }
}

// =====================================================================
// Kernel 3: gather + BoxCoder.decode + clip_to_image + det score.
// =====================================================================
__global__ __launch_bounds__(256)
void decode_kernel(const int* __restrict__ sel_idx,         // [N*PRE_TOPN]
                   const float* __restrict__ masked,        // [N*HW]
                   const float* __restrict__ box_reg,       // [N,4,HW]
                   const float* __restrict__ anchors,       // [N,HW,4]
                   float* __restrict__ boxes,               // [N*PRE_TOPN,4]
                   float* __restrict__ dets)                // [N*PRE_TOPN]
{
    const int t = blockIdx.x * blockDim.x + threadIdx.x;
    if (t >= NIMG * PRE_TOPN) return;
    const int n = t / PRE_TOPN;
    const int idx = sel_idx[t];

    float4 out = make_float4(0.f, 0.f, 0.f, 0.f);
    float det = 0.0f;
    if (idx >= 0) {
        const int hw = idx;
        const float v = masked[(size_t)n * HW_ + hw];
        det = (v > 0.0f) ? sqrtf(v) : 0.0f;

        const float* rr = box_reg + (size_t)n * 4 * HW_ + hw;
        const float dx = rr[0 * HW_] * 0.1f;
        const float dy = rr[1 * HW_] * 0.1f;
        const float dw = fminf(rr[2 * HW_] * 0.2f, DWH_CLIP_);
        const float dh = fminf(rr[3 * HW_] * 0.2f, DWH_CLIP_);

        const float* ap = anchors + ((size_t)n * HW_ + hw) * 4;
        const float x1 = ap[0], y1 = ap[1], x2 = ap[2], y2 = ap[3];
        const float w  = x2 - x1 + 1.0f;
        const float h  = y2 - y1 + 1.0f;
        const float cx = x1 + 0.5f * w;
        const float cy = y1 + 0.5f * h;
        const float pcx = dx * w + cx;
        const float pcy = dy * h + cy;
        const float pw  = __builtin_amdgcn_exp2f(dw * LOG2E_) * w;
        const float ph  = __builtin_amdgcn_exp2f(dh * LOG2E_) * h;

        out.x = fminf(fmaxf(pcx - 0.5f * pw,        0.f), IMG_W_ - 1.f);
        out.y = fminf(fmaxf(pcy - 0.5f * ph,        0.f), IMG_H_ - 1.f);
        out.z = fminf(fmaxf(pcx + 0.5f * pw - 1.f,  0.f), IMG_W_ - 1.f);
        out.w = fminf(fmaxf(pcy + 0.5f * ph - 1.f,  0.f), IMG_H_ - 1.f);
    }
    ((float4*)boxes)[t] = out;
    dets[t] = det;
}

// =====================================================================
// Kernel 4: greedy NMS, 100 fixed iterations, one workgroup per image.
// Boxes + scores held in LDS (~28 KB of the WGP's 320 KB).
// Tie-break on argmax matches jnp.argmax (lowest index wins).
// =====================================================================
__global__ __launch_bounds__(1024)
void nms_kernel(const float* __restrict__ boxes,            // [N,PRE_TOPN,4]
                const float* __restrict__ dets,             // [N,PRE_TOPN]
                float* __restrict__ out)                    // boxes(3200) ++ scores(800)
{
    __shared__ float sb[PRE_TOPN * 4];
    __shared__ float ss[PRE_TOPN];
    __shared__ float redV[1024];
    __shared__ int   redI[1024];

    const int n   = blockIdx.x;
    const int tid = threadIdx.x;

    if (tid < PRE_TOPN) {
        ((float4*)sb)[tid] = ((const float4*)boxes)[n * PRE_TOPN + tid];
        ss[tid] = dets[n * PRE_TOPN + tid];
    }
    __syncthreads();

    float* kb  = out + (size_t)n * POST_TOPN * 4;
    float* ksc = out + (size_t)NIMG * POST_TOPN * 4 + (size_t)n * POST_TOPN;

    for (int it = 0; it < POST_TOPN; ++it) {
        redV[tid] = (tid < PRE_TOPN) ? ss[tid] : -1.0f;
        redI[tid] = tid;
        __syncthreads();
        for (int st = 512; st > 0; st >>= 1) {
            if (tid < st) {
                const float v1 = redV[tid],      v2 = redV[tid + st];
                const int   i1 = redI[tid],      i2 = redI[tid + st];
                if (v2 > v1 || (v2 == v1 && i2 < i1)) { redV[tid] = v2; redI[tid] = i2; }
            }
            __syncthreads();
        }
        const int   j    = redI[0];
        const float sj   = redV[0];
        const float flag = (sj > 0.0f) ? 1.0f : 0.0f;
        const float bx1 = sb[j * 4 + 0], by1 = sb[j * 4 + 1];
        const float bx2 = sb[j * 4 + 2], by2 = sb[j * 4 + 3];

        if (tid == 0) {
            kb[it * 4 + 0] = bx1 * flag;
            kb[it * 4 + 1] = by1 * flag;
            kb[it * 4 + 2] = bx2 * flag;
            kb[it * 4 + 3] = by2 * flag;
            ksc[it] = sj * flag;
        }

        if (tid < PRE_TOPN) {
            const float x1 = sb[tid * 4 + 0], y1 = sb[tid * 4 + 1];
            const float x2 = sb[tid * 4 + 2], y2 = sb[tid * 4 + 3];
            const float a1 = (bx2 - bx1 + 1.0f) * (by2 - by1 + 1.0f);
            const float a2 = (x2 - x1 + 1.0f) * (y2 - y1 + 1.0f);
            const float lx = fmaxf(bx1, x1), ly = fmaxf(by1, y1);
            const float rx = fminf(bx2, x2), ry = fminf(by2, y2);
            const float iw = fmaxf(rx - lx + 1.0f, 0.0f);
            const float ih = fmaxf(ry - ly + 1.0f, 0.0f);
            const float inter = iw * ih;
            const float iou = inter * frcp(a1 + a2 - inter);  // v_rcp_f32, not IEEE div
            if (iou > NMS_TH || tid == j) ss[tid] = 0.0f;     // suppression applies
        }                                                     // even when flag==0
        __syncthreads();
    }
}

// =====================================================================
// launch
// =====================================================================
extern "C" void kernel_launch(void* const* d_in, const int* in_sizes, int n_in,
                              void* d_out, int out_size, void* d_ws, size_t ws_size,
                              hipStream_t stream) {
    const float* box_regression = (const float*)d_in[0];  // [8,4,100,152]
    const float* centerness     = (const float*)d_in[1];  // [8,1,100,152] == [N*HW]
    const float* anchors        = (const float*)d_in[2];  // [8,15200,4]
    const float* logits         = (const float*)d_in[3];  // [8,15200,256]
    float* out = (float*)d_out;                           // 3200 box + 800 score floats

    // workspace layout (~678 KB total, 16B-aligned sections)
    char* ws = (char*)d_ws;
    float* masked = (float*)(ws);                                   // N*HW floats
    int*   sel    = (int*)  (ws + (size_t)NIMG * HW_ * 4);          // N*1000 ints
    float* boxes  = (float*)(ws + (size_t)NIMG * HW_ * 4 + 32000);  // N*1000*4 floats
    float* dets   = (float*)(ws + (size_t)NIMG * HW_ * 4 + 32000 + 128000); // N*1000

    // 121600 rows / 16 rows-per-wave / 8 waves-per-block = 950 blocks
    scores_wmma_kernel<<<950, 256, 0, stream>>>(logits, centerness, masked);
    topk_select_kernel<<<NIMG, 1024, 0, stream>>>(masked, sel);
    decode_kernel<<<(NIMG * PRE_TOPN + 255) / 256, 256, 0, stream>>>(
        sel, masked, box_regression, anchors, boxes, dets);
    nms_kernel<<<NIMG, 1024, 0, stream>>>(boxes, dets, out);
}
